// CellFoundationSelfAttention_833223656372
// MI455X (gfx1250) — matrix operand order repaired and checked by reading, not compiled, not run on hardware
//
#include <hip/hip_runtime.h>
#include <hip/hip_bf16.h>

typedef __attribute__((ext_vector_type(16))) _Float16 v16h;
typedef __attribute__((ext_vector_type(8)))  _Float16 v8h;
typedef __attribute__((ext_vector_type(4)))  _Float16 v4h;
typedef __attribute__((ext_vector_type(8)))  float    v8f;

#define BB 4
#define SS 1024
#define HID 1024
#define NH 16
#define DH 64

static __device__ __forceinline__ v16h ld16(const _Float16* p) {
  return *(const v16h*)p;
}
// assemble an A-fragment from two contiguous 8-half (16B) chunks
static __device__ __forceinline__ v16h cat8(const _Float16* p0, const _Float16* p1) {
  v8h lo = *(const v8h*)p0;
  v8h hi = *(const v8h*)p1;
  return __builtin_shufflevector(lo, hi, 0,1,2,3,4,5,6,7,8,9,10,11,12,13,14,15);
}
static __device__ __forceinline__ v8f wmma_f16(v16h a, v16h b, v8f c) {
  return __builtin_amdgcn_wmma_f32_16x16x32_f16(false, a, false, b, (short)0, c, false, false);
}

// ---------------------------------------------------------------------------
// Kernel 0: fp32 -> f16 (hidden states), vectorized x4
// ---------------------------------------------------------------------------
__global__ __launch_bounds__(256) void cvt_f16_kernel(const float* __restrict__ in,
                                                      _Float16* __restrict__ out, int n) {
  int i = (blockIdx.x * blockDim.x + threadIdx.x) * 4;
  if (i >= n) return;
  float4 v = *(const float4*)(in + i);
  v4h o;
  o.x = (_Float16)v.x; o.y = (_Float16)v.y; o.z = (_Float16)v.z; o.w = (_Float16)v.w;
  *(v4h*)(out + i) = o;
}

// ---------------------------------------------------------------------------
// Kernel 1: W (K,N) fp32 row-major -> Wt (N,K) f16 row-major, LDS-tiled
// ---------------------------------------------------------------------------
__global__ __launch_bounds__(256) void transpose_w_f16_kernel(const float* __restrict__ in,
                                                              _Float16* __restrict__ out) {
  __shared__ _Float16 tile[32][33];
  int tx = threadIdx.x, ty = threadIdx.y;
  int nbase = blockIdx.x * 32;
  int kbase = blockIdx.y * 32;
#pragma unroll
  for (int j = ty; j < 32; j += 8)
    tile[j][tx] = (_Float16)in[(size_t)(kbase + j) * HID + nbase + tx];
  __syncthreads();
#pragma unroll
  for (int j = ty; j < 32; j += 8)
    out[(size_t)(nbase + j) * HID + kbase + tx] = tile[tx][j];
}

// ---------------------------------------------------------------------------
// Kernel 2: fused QKV projection + bias + RoPE.
//   grid = (M/64, 3*NH); block = 128 (4 waves), wave -> 16x64 output tile.
//   Q,K written f16 (B,H,S,DH); V written transposed f16 (B,H,DH,S).
// ---------------------------------------------------------------------------
__global__ __launch_bounds__(128) void qkv_rope_kernel(
    const _Float16* __restrict__ X,   // (B*S, HID) f16
    const _Float16* __restrict__ Wt,  // (3, HID(N), HID(K)) f16
    const float* __restrict__ bq, const float* __restrict__ bk, const float* __restrict__ bv,
    const float* __restrict__ freqs,  // (S, 64) fp32
    _Float16* __restrict__ Qo, _Float16* __restrict__ Ko, _Float16* __restrict__ Vto) {
  int w    = threadIdx.x >> 5;
  int lane = threadIdx.x & 31;
  int lh = lane >> 4, ln = lane & 15;
  int mbase = blockIdx.x * 64 + w * 16;
  int nb = blockIdx.y;
  int type = nb >> 4;       // 0=Q 1=K 2=V
  int h    = nb & 15;
  const _Float16* Wtt = Wt + (size_t)type * HID * HID;
  const _Float16* xr  = X + (size_t)(mbase + ln) * HID;

  v8f acc0 = {}, acc1 = {}, acc2 = {}, acc3 = {};
  for (int kk = 0; kk < HID; kk += 32) {
    v16h a = cat8(xr + kk + lh * 16, xr + kk + lh * 16 + 8);
    v16h b0 = ld16(Wtt + (size_t)(h * 64 +  0 + ln) * HID + kk + lh * 16);
    v16h b1 = ld16(Wtt + (size_t)(h * 64 + 16 + ln) * HID + kk + lh * 16);
    v16h b2 = ld16(Wtt + (size_t)(h * 64 + 32 + ln) * HID + kk + lh * 16);
    v16h b3 = ld16(Wtt + (size_t)(h * 64 + 48 + ln) * HID + kk + lh * 16);
    acc0 = wmma_f16(a, b0, acc0);
    acc1 = wmma_f16(a, b1, acc1);
    acc2 = wmma_f16(a, b2, acc2);
    acc3 = wmma_f16(a, b3, acc3);
  }

  const float* bias = (type == 0) ? bq : (type == 1) ? bk : bv;
  float bb0 = bias[h * 64 +  0 + ln];
  float bb1 = bias[h * 64 + 16 + ln];
  float bb2 = bias[h * 64 + 32 + ln];
  float bb3 = bias[h * 64 + 48 + ln];
#pragma unroll
  for (int r = 0; r < 8; ++r) {
    acc0[r] += bb0; acc1[r] += bb1; acc2[r] += bb2; acc3[r] += bb3;
  }

  if (type < 2) {
    _Float16* O = (type == 0) ? Qo : Ko;
#pragma unroll
    for (int r = 0; r < 8; ++r) {
      int m    = mbase + r + lh * 8;
      int bidx = m >> 10;
      int spos = m & (SS - 1);
      _Float16* orow = O + ((size_t)(bidx * NH + h) * SS + spos) * DH;
      // pair (tile0, tile2): d = ln, d+32 ; pair (tile1, tile3): d = 16+ln, d+32
      float f0 = freqs[spos * DH + ln];
      float f1 = freqs[spos * DH + 16 + ln];
      float c0, s0; __sincosf(f0, &s0, &c0);
      float c1, s1; __sincosf(f1, &s1, &c1);
      float x1 = acc0[r], x2 = acc2[r];
      orow[ln]      = (_Float16)(x1 * c0 - x2 * s0);
      orow[32 + ln] = (_Float16)(x2 * c0 + x1 * s0);
      x1 = acc1[r]; x2 = acc3[r];
      orow[16 + ln] = (_Float16)(x1 * c1 - x2 * s1);
      orow[48 + ln] = (_Float16)(x2 * c1 + x1 * s1);
    }
  } else {
#pragma unroll
    for (int r = 0; r < 8; ++r) {
      int m    = mbase + r + lh * 8;
      int bidx = m >> 10;
      int spos = m & (SS - 1);
      _Float16* obh = Vto + (size_t)(bidx * NH + h) * DH * SS;
      obh[(size_t)( 0 + ln) * SS + spos] = (_Float16)acc0[r];
      obh[(size_t)(16 + ln) * SS + spos] = (_Float16)acc1[r];
      obh[(size_t)(32 + ln) * SS + spos] = (_Float16)acc2[r];
      obh[(size_t)(48 + ln) * SS + spos] = (_Float16)acc3[r];
    }
  }
}

// ---------------------------------------------------------------------------
// Kernel 3: flash-attention. grid=(S/64, NH, B), block=128 (4 waves),
// wave -> 16 query rows, loops over 32-key blocks.
// ---------------------------------------------------------------------------
__global__ __launch_bounds__(128) void attn_kernel(
    const _Float16* __restrict__ Q,   // (B,H,S,DH) f16
    const _Float16* __restrict__ K,   // (B,H,S,DH) f16
    const _Float16* __restrict__ Vt,  // (B,H,DH,S) f16
    const float* __restrict__ mask,   // (B,S)
    float* __restrict__ out) {        // (B,S,HID)
  __shared__ __align__(32) _Float16 plds[4][16 * 32];
  int w    = threadIdx.x >> 5;
  int lane = threadIdx.x & 31;
  int lh = lane >> 4, ln = lane & 15;
  int b = blockIdx.z, h = blockIdx.y;
  int qbase = blockIdx.x * 64 + w * 16;

  const _Float16* Qbh = Q  + (size_t)(b * NH + h) * SS * DH;
  const _Float16* Kbh = K  + (size_t)(b * NH + h) * SS * DH;
  const _Float16* Vbh = Vt + (size_t)(b * NH + h) * DH * SS;
  const float*   mrow = mask + b * SS;
  _Float16* pl = plds[w];

  // Q A-fragments (K-dim chunks d 0..31 and 32..63)
  const _Float16* qp = Qbh + (size_t)(qbase + ln) * DH;
  v16h a0 = cat8(qp + lh * 16,      qp + lh * 16 + 8);
  v16h a1 = cat8(qp + 32 + lh * 16, qp + 32 + lh * 16 + 8);

  v8f acc0 = {}, acc1 = {}, acc2 = {}, acc3 = {};
  float m_[8], l_[8];
#pragma unroll
  for (int r = 0; r < 8; ++r) { m_[r] = -1e30f; l_[r] = 0.0f; }

  for (int kb = 0; kb < SS; kb += 32) {
    // K B-fragments: N = key, K-dim = d
    const _Float16* k0 = Kbh + (size_t)(kb + ln) * DH;
    const _Float16* k1 = Kbh + (size_t)(kb + 16 + ln) * DH;
    v16h bk00 = ld16(k0 + lh * 16);
    v16h bk01 = ld16(k0 + 32 + lh * 16);
    v16h bk10 = ld16(k1 + lh * 16);
    v16h bk11 = ld16(k1 + 32 + lh * 16);
    v8f s0 = {}, s1 = {};
    s0 = wmma_f16(a0, bk00, s0);
    s0 = wmma_f16(a1, bk01, s0);
    s1 = wmma_f16(a0, bk10, s1);
    s1 = wmma_f16(a1, bk11, s1);

    float mv0 = mrow[kb + ln];
    float mv1 = mrow[kb + 16 + ln];
#pragma unroll
    for (int r = 0; r < 8; ++r) {
      float x0 = s0[r] * 0.125f + mv0;
      float x1 = s1[r] * 0.125f + mv1;
      // row max across the 16-lane half (N dimension)
      float v = fmaxf(x0, x1);
      v = fmaxf(v, __shfl_xor(v, 1, 32));
      v = fmaxf(v, __shfl_xor(v, 2, 32));
      v = fmaxf(v, __shfl_xor(v, 4, 32));
      v = fmaxf(v, __shfl_xor(v, 8, 32));
      float mn = fmaxf(m_[r], v);
      float alpha = __expf(m_[r] - mn);
      float p0 = __expf(x0 - mn);
      float p1 = __expf(x1 - mn);
      float rs = p0 + p1;
      rs += __shfl_xor(rs, 1, 32);
      rs += __shfl_xor(rs, 2, 32);
      rs += __shfl_xor(rs, 4, 32);
      rs += __shfl_xor(rs, 8, 32);
      l_[r] = l_[r] * alpha + rs;
      m_[r] = mn;
      acc0[r] *= alpha; acc1[r] *= alpha; acc2[r] *= alpha; acc3[r] *= alpha;
      int row = r + lh * 8;
      pl[row * 32 + ln]      = (_Float16)p0;
      pl[row * 32 + 16 + ln] = (_Float16)p1;
    }
    // re-swizzle probs C-layout -> A-layout via LDS (wave-in-order DS ops)
    v16h pa = cat8(pl + ln * 32 + lh * 16, pl + ln * 32 + lh * 16 + 8);
    // V B-fragments: N = d (column of ctx), K-dim = keys; Vt rows are contiguous keys
    v16h bv0 = ld16(Vbh + (size_t)( 0 + ln) * SS + kb + lh * 16);
    v16h bv1 = ld16(Vbh + (size_t)(16 + ln) * SS + kb + lh * 16);
    v16h bv2 = ld16(Vbh + (size_t)(32 + ln) * SS + kb + lh * 16);
    v16h bv3 = ld16(Vbh + (size_t)(48 + ln) * SS + kb + lh * 16);
    acc0 = wmma_f16(pa, bv0, acc0);
    acc1 = wmma_f16(pa, bv1, acc1);
    acc2 = wmma_f16(pa, bv2, acc2);
    acc3 = wmma_f16(pa, bv3, acc3);
  }

  size_t obase = ((size_t)b * SS) * HID + h * DH;
#pragma unroll
  for (int r = 0; r < 8; ++r) {
    float inv = 1.0f / l_[r];
    int q = qbase + r + lh * 8;
    float* orow = out + obase + (size_t)q * HID;
    orow[ 0 + ln] = acc0[r] * inv;
    orow[16 + ln] = acc1[r] * inv;
    orow[32 + ln] = acc2[r] * inv;
    orow[48 + ln] = acc3[r] * inv;
  }
}

// ---------------------------------------------------------------------------
extern "C" void kernel_launch(void* const* d_in, const int* in_sizes, int n_in,
                              void* d_out, int out_size, void* d_ws, size_t ws_size,
                              hipStream_t stream) {
  const float* hidden = (const float*)d_in[0];
  const float* mask   = (const float*)d_in[1];
  const float* freqs  = (const float*)d_in[2];
  const float* Wq     = (const float*)d_in[3];
  const float* bq     = (const float*)d_in[4];
  const float* Wk     = (const float*)d_in[5];
  const float* bk     = (const float*)d_in[6];
  const float* Wv     = (const float*)d_in[7];
  const float* bv     = (const float*)d_in[8];
  float* out = (float*)d_out;
  char* ws = (char*)d_ws;

  const size_t MB = 1024ull * 1024ull;
  _Float16* Xh  = (_Float16*)(ws + 0);        //  8 MB: hidden f16 (4096,1024)
  _Float16* Wt  = (_Float16*)(ws + 8 * MB);   //  6 MB: Wt f16 (3,1024,1024) N-major
  _Float16* Qh  = (_Float16*)(ws + 14 * MB);  //  8 MB: Q  f16 (B,H,S,DH)
  _Float16* Kh  = (_Float16*)(ws + 22 * MB);  //  8 MB: K  f16 (B,H,S,DH)
  _Float16* Vth = (_Float16*)(ws + 30 * MB);  //  8 MB: Vt f16 (B,H,DH,S)

  // 0) hidden fp32 -> f16
  int nhid = BB * SS * HID;
  cvt_f16_kernel<<<nhid / (256 * 4), 256, 0, stream>>>(hidden, Xh, nhid);
  // 1) weights fp32 (K,N) -> f16 (N,K)
  dim3 tg(HID / 32, HID / 32), tb(32, 8);
  transpose_w_f16_kernel<<<tg, tb, 0, stream>>>(Wq, Wt);
  transpose_w_f16_kernel<<<tg, tb, 0, stream>>>(Wk, Wt + (size_t)HID * HID);
  transpose_w_f16_kernel<<<tg, tb, 0, stream>>>(Wv, Wt + 2 * (size_t)HID * HID);
  // 2) QKV + bias + RoPE
  qkv_rope_kernel<<<dim3(BB * SS / 64, 3 * NH), 128, 0, stream>>>(
      Xh, Wt, bq, bk, bv, freqs, Qh, Kh, Vth);
  // 3) flash attention
  attn_kernel<<<dim3(SS / 64, NH, BB), 128, 0, stream>>>(Qh, Kh, Vth, mask, out);
}
